// MoleculeGNN_29025388986465
// MI455X (gfx1250) — compile-verified, matrix operand order checked by reading, not run on hardware
//
#include <hip/hip_runtime.h>
#include <hip/hip_bf16.h>

typedef __bf16 bf16_t;
typedef __attribute__((ext_vector_type(16))) __bf16 v16bf;
typedef __attribute__((ext_vector_type(8)))  __bf16 v8bf;
typedef __attribute__((ext_vector_type(8)))  float  v8f;

static inline long cdivl(long a, long b) { return (a + b - 1) / b; }

// ---------------------------------------------------------------------------
// f32 -> bf16 conversion (activations)
// ---------------------------------------------------------------------------
__global__ __launch_bounds__(256) void cvt_f32_to_bf16(const float* __restrict__ in,
                                                       bf16_t* __restrict__ out, long n) {
    long i = (long)blockIdx.x * blockDim.x + threadIdx.x;
    if (i < n) out[i] = (bf16_t)in[i];
}

// ---------------------------------------------------------------------------
// fill
// ---------------------------------------------------------------------------
__global__ __launch_bounds__(256) void fill_f32(float* __restrict__ p, float v, long n) {
    long i = (long)blockIdx.x * blockDim.x + threadIdx.x;
    if (i < n) p[i] = v;
}

// ---------------------------------------------------------------------------
// Pack B (f32 weight [K, Nc], row-major) into WMMA bf16 fragment order.
// Fragment layout for lane l (ISA 7.12.2, 16-bit 32x16 B, mirrored from A):
//   col = l&15, kbase = (l<16)?0:8
//   element j: k = kt*32 + kbase + j       (j<8)
//              k = kt*32 + kbase + 8 + j   (j>=8)
// Stored at Bp[(((nt*KT + kt)*32) + l)*16 + j]  -> one contiguous 32B v16bf/lane.
// One thread per (nt, kt, lane).
// ---------------------------------------------------------------------------
__global__ __launch_bounds__(256) void pack_b_frag(const float* __restrict__ W,
                                                   bf16_t* __restrict__ Bp,
                                                   int K, int Nc) {
    const int KT = K >> 5;            // k-tiles of 32
    const int NT = Nc >> 4;           // n-tiles of 16
    int t = blockIdx.x * blockDim.x + threadIdx.x;
    if (t >= NT * KT * 32) return;
    int lane = t & 31;
    int kt   = (t >> 5) % KT;
    int nt   = (t >> 5) / KT;
    const int col   = nt * 16 + (lane & 15);
    const int kbase = kt * 32 + ((lane < 16) ? 0 : 8);
    bf16_t* dst = Bp + (size_t)t * 16;
#pragma unroll
    for (int j = 0; j < 16; ++j) {
        const int kk = kbase + ((j < 8) ? j : (8 + j));
        dst[j] = (bf16_t)W[(long)kk * Nc + col];
    }
}

// ---------------------------------------------------------------------------
// WMMA bf16 GEMM: C[M,Nc] = A[M,K] @ B[K,Nc], f32 accumulate.
// One wave per 16x16 C tile, 8 waves (256 threads) per block.
// A: row-major bf16; per lane two contiguous 16B chunks -> shuffle into v16bf.
// B: pre-packed fragments (pack_b_frag) -> single 32B v16bf load per lane/iter.
// ---------------------------------------------------------------------------
__global__ __launch_bounds__(256) void gemm_bf16_wmma(const bf16_t* __restrict__ A,
                                                      const bf16_t* __restrict__ Bp,
                                                      float* __restrict__ C,
                                                      int M, int K, int Nc) {
    const int wave = threadIdx.x >> 5;
    const int lane = threadIdx.x & 31;
    const int tileN = blockIdx.x;
    const int tileM = blockIdx.y * 8 + wave;
    const int row0 = tileM * 16;
    if (row0 + 16 > M) return;

    const int KT    = K >> 5;
    const int arow  = row0 + (lane & 15);
    const int kbase = (lane < 16) ? 0 : 8;
    const bf16_t* __restrict__ Arow = A + (long)arow * K + kbase;
    const v16bf* __restrict__ Bfrag =
        (const v16bf*)Bp + ((long)tileN * KT) * 32 + lane;

    v8f acc = {};
#pragma unroll 4
    for (int kt = 0; kt < KT; ++kt) {
        const v8bf a_lo = *(const v8bf*)(Arow + kt * 32);
        const v8bf a_hi = *(const v8bf*)(Arow + kt * 32 + 16);
        const v16bf af = __builtin_shufflevector(a_lo, a_hi,
            0, 1, 2, 3, 4, 5, 6, 7, 8, 9, 10, 11, 12, 13, 14, 15);
        const v16bf bf = Bfrag[(long)kt * 32];
        acc = __builtin_amdgcn_wmma_f32_16x16x32_bf16(
            /*neg_a=*/false, af, /*neg_b=*/false, bf,
            /*c_mod=*/(short)0, acc, /*reuse_a=*/false, /*reuse_b=*/false);
    }

    const int bcol  = tileN * 16 + (lane & 15);
    const int mbase = row0 + ((lane < 16) ? 0 : 8);
#pragma unroll
    for (int r = 0; r < 8; ++r) {
        C[(long)(mbase + r) * Nc + bcol] = acc[r];
    }
}

// ---------------------------------------------------------------------------
// al_s[n,h] = dot(h[n,h,:], a_src[h,:]);  al_d likewise
// ---------------------------------------------------------------------------
__global__ __launch_bounds__(256) void attn_coef(const float* __restrict__ h,
                                                 const float* __restrict__ a_src,
                                                 const float* __restrict__ a_dst,
                                                 float* __restrict__ als,
                                                 float* __restrict__ ald,
                                                 int N, int H, int C) {
    int idx = blockIdx.x * blockDim.x + threadIdx.x;
    if (idx >= N * H) return;
    int n = idx / H, hd = idx - n * H;
    const float* hp = h + (long)n * H * C + (long)hd * C;
    const float* as = a_src + (long)hd * C;
    const float* ad = a_dst + (long)hd * C;
    float s = 0.f, d = 0.f;
    for (int c = 0; c < C; ++c) { s += hp[c] * as[c]; d += hp[c] * ad[c]; }
    als[idx] = s;
    ald[idx] = d;
}

// ---------------------------------------------------------------------------
// float atomic max via ordered-int trick (works for mixed signs, init -FLT_MAX)
// ---------------------------------------------------------------------------
__device__ inline void atomicMaxFloat(float* addr, float v) {
    if (v >= 0.0f)
        atomicMax(reinterpret_cast<int*>(addr), __float_as_int(v));
    else
        atomicMin(reinterpret_cast<unsigned int*>(addr), __float_as_uint(v));
}

// ---------------------------------------------------------------------------
// pass 1: e = leaky_relu(al_s[src] + al_d[dst]); segment max over dst
// edge index e < E: real edge; e >= E: self loop (src=dst=e-E)
// ---------------------------------------------------------------------------
__global__ __launch_bounds__(256) void edge_logits(const int* __restrict__ ei,
                                                   int E, int Etot, int H,
                                                   const float* __restrict__ als,
                                                   const float* __restrict__ ald,
                                                   float* __restrict__ ebuf,
                                                   float* __restrict__ mbuf) {
    int t = blockIdx.x * blockDim.x + threadIdx.x;
    if (t >= Etot * H) return;
    int e = t / H, hd = t - e * H;
    int src, dst;
    if (e < E) { src = ei[e]; dst = ei[E + e]; } else { src = dst = e - E; }
    float v = als[src * H + hd] + ald[dst * H + hd];
    v = (v > 0.f) ? v : 0.2f * v;
    ebuf[t] = v;
    atomicMaxFloat(&mbuf[dst * H + hd], v);
}

// ---------------------------------------------------------------------------
// pass 2: ex = exp(e - m[dst]); segment sum over dst (in-place over ebuf)
// ---------------------------------------------------------------------------
__global__ __launch_bounds__(256) void edge_expsum(const int* __restrict__ ei,
                                                   int E, int Etot, int H,
                                                   float* __restrict__ ebuf,
                                                   const float* __restrict__ mbuf,
                                                   float* __restrict__ dbuf) {
    int t = blockIdx.x * blockDim.x + threadIdx.x;
    if (t >= Etot * H) return;
    int e = t / H, hd = t - e * H;
    int dst = (e < E) ? ei[E + e] : (e - E);
    float ex = expf(ebuf[t] - mbuf[dst * H + hd]);
    ebuf[t] = ex;
    atomicAdd(&dbuf[dst * H + hd], ex);
}

// ---------------------------------------------------------------------------
// pass 3: out[dst] += alpha * h[src];  one thread per (edge, 4 channels)
// ---------------------------------------------------------------------------
__global__ __launch_bounds__(256) void edge_agg(const int* __restrict__ ei,
                                                int E, int Etot, int H, int C,
                                                const float* __restrict__ ebuf,
                                                const float* __restrict__ dbuf,
                                                const float* __restrict__ h,
                                                float* __restrict__ out) {
    const int HC = H * C;
    const int gpe = HC >> 2;  // groups of 4 channels per edge
    long t = (long)blockIdx.x * blockDim.x + threadIdx.x;
    if (t >= (long)Etot * gpe) return;
    int e = (int)(t / gpe);
    int g = (int)(t - (long)e * gpe);
    int ch0 = g << 2;
    int hd = ch0 / C;  // C is a multiple of 4, so head is uniform in the group
    int src, dst;
    if (e < E) { src = ei[e]; dst = ei[E + e]; } else { src = dst = e - E; }
    float alpha = ebuf[(long)e * H + hd] / (dbuf[dst * H + hd] + 1e-16f);
    const float4 hv = *reinterpret_cast<const float4*>(h + (long)src * HC + ch0);
    float* od = out + (long)dst * HC + ch0;
    atomicAdd(&od[0], alpha * hv.x);
    atomicAdd(&od[1], alpha * hv.y);
    atomicAdd(&od[2], alpha * hv.z);
    atomicAdd(&od[3], alpha * hv.w);
}

// ---------------------------------------------------------------------------
// out = (elu?)(in + bias)
// ---------------------------------------------------------------------------
__global__ __launch_bounds__(256) void bias_act(const float* __restrict__ in,
                                                const float* __restrict__ bias,
                                                float* __restrict__ out,
                                                long n, int HC, int do_elu) {
    long t = (long)blockIdx.x * blockDim.x + threadIdx.x;
    if (t >= n) return;
    int c = (int)(t % HC);
    float v = in[t] + bias[c];
    if (do_elu) v = (v > 0.f) ? v : (expf(v) - 1.0f);
    out[t] = v;
}

// ---------------------------------------------------------------------------
// global mean pool
// ---------------------------------------------------------------------------
__global__ __launch_bounds__(256) void pool_acc(const float* __restrict__ h,
                                                const int* __restrict__ batch,
                                                float* __restrict__ sums,
                                                float* __restrict__ cnt,
                                                int N, int C) {
    int t = blockIdx.x * blockDim.x + threadIdx.x;
    if (t >= N * C) return;
    int n = t / C, c = t - n * C;
    int g = batch[n];
    atomicAdd(&sums[g * C + c], h[t]);
    if (c == 0) atomicAdd(&cnt[g], 1.0f);
}

__global__ __launch_bounds__(256) void pool_div(const float* __restrict__ sums,
                                                const float* __restrict__ cnt,
                                                float* __restrict__ out,
                                                int G, int C) {
    int t = blockIdx.x * blockDim.x + threadIdx.x;
    if (t >= G * C) return;
    out[t] = sums[t] / fmaxf(cnt[t / C], 1.0f);
}

// ---------------------------------------------------------------------------
// one GAT layer
// ---------------------------------------------------------------------------
static void run_layer(const float* act, int K, const float* W,
                      const float* a_s, const float* a_d, const float* bias,
                      int H, int C, int N, const int* ei, int E, int Etot,
                      bf16_t* xb, bf16_t* Bp, float* hbuf,
                      float* als, float* ald, float* mbuf, float* dbuf,
                      float* ebuf, float* agg, float* out_act, bool do_elu,
                      hipStream_t stream) {
    const int HC = H * C;
    const long nx = (long)N * K;
    cvt_f32_to_bf16<<<(unsigned)cdivl(nx, 256), 256, 0, stream>>>(act, xb, nx);

    const int npack = (HC / 16) * (K / 32) * 32;
    pack_b_frag<<<(unsigned)cdivl(npack, 256), 256, 0, stream>>>(W, Bp, K, HC);

    dim3 ggrid((unsigned)(HC / 16), (unsigned)cdivl(cdivl(N, 16), 8));
    gemm_bf16_wmma<<<ggrid, 256, 0, stream>>>(xb, Bp, hbuf, N, K, HC);

    const long nh = (long)N * H;
    attn_coef<<<(unsigned)cdivl(nh, 256), 256, 0, stream>>>(hbuf, a_s, a_d, als, ald, N, H, C);
    fill_f32<<<(unsigned)cdivl(nh, 256), 256, 0, stream>>>(mbuf, -3.0e38f, nh);
    fill_f32<<<(unsigned)cdivl(nh, 256), 256, 0, stream>>>(dbuf, 0.0f, nh);
    const long na = (long)N * HC;
    fill_f32<<<(unsigned)cdivl(na, 256), 256, 0, stream>>>(agg, 0.0f, na);

    const long ne = (long)Etot * H;
    edge_logits<<<(unsigned)cdivl(ne, 256), 256, 0, stream>>>(ei, E, Etot, H, als, ald, ebuf, mbuf);
    edge_expsum<<<(unsigned)cdivl(ne, 256), 256, 0, stream>>>(ei, E, Etot, H, ebuf, mbuf, dbuf);
    const long nagg = (long)Etot * (HC / 4);
    edge_agg<<<(unsigned)cdivl(nagg, 256), 256, 0, stream>>>(ei, E, Etot, H, C, ebuf, dbuf, hbuf, agg);

    bias_act<<<(unsigned)cdivl(na, 256), 256, 0, stream>>>(agg, bias, out_act, na, HC, do_elu ? 1 : 0);
}

// ---------------------------------------------------------------------------
// entry point
// ---------------------------------------------------------------------------
extern "C" void kernel_launch(void* const* d_in, const int* in_sizes, int n_in,
                              void* d_out, int out_size, void* d_ws, size_t ws_size,
                              hipStream_t stream) {
    const float* x     = (const float*)d_in[0];
    const int*   ei    = (const int*)d_in[1];
    const int*   batch = (const int*)d_in[2];
    const float* W1  = (const float*)d_in[3];
    const float* as1 = (const float*)d_in[4];
    const float* ad1 = (const float*)d_in[5];
    const float* b1  = (const float*)d_in[6];
    const float* W2  = (const float*)d_in[7];
    const float* as2 = (const float*)d_in[8];
    const float* ad2 = (const float*)d_in[9];
    const float* b2  = (const float*)d_in[10];
    const float* W3  = (const float*)d_in[11];
    const float* as3 = (const float*)d_in[12];
    const float* ad3 = (const float*)d_in[13];
    const float* b3  = (const float*)d_in[14];

    const int F_IN = 64, HID = 64, OUT_C = 64, HEADS = 4;
    const int N = in_sizes[0] / F_IN;     // 50000
    const int E = in_sizes[1] / 2;        // 800000
    const int Etot = E + N;               // + self loops
    const int G = out_size / OUT_C;       // 256
    const int HCmax = HEADS * HID;        // 256

    // workspace carve (256B aligned)
    char* p = (char*)d_ws;
    auto take = [&](size_t bytes) -> char* {
        char* r = p;
        p += (bytes + 255) & ~(size_t)255;
        return r;
    };
    float*  actA = (float*)take((size_t)N * HCmax * sizeof(float));
    float*  agg  = (float*)take((size_t)N * HCmax * sizeof(float));
    float*  hbuf = (float*)take((size_t)N * HCmax * sizeof(float));
    bf16_t* xb   = (bf16_t*)take((size_t)N * HCmax * sizeof(bf16_t));
    bf16_t* Bp   = (bf16_t*)take((size_t)HCmax * HCmax * sizeof(bf16_t));
    float*  als  = (float*)take((size_t)N * HEADS * sizeof(float));
    float*  ald  = (float*)take((size_t)N * HEADS * sizeof(float));
    float*  mbuf = (float*)take((size_t)N * HEADS * sizeof(float));
    float*  dbuf = (float*)take((size_t)N * HEADS * sizeof(float));
    float*  ebuf = (float*)take((size_t)Etot * HEADS * sizeof(float));
    float*  pool = (float*)take((size_t)G * OUT_C * sizeof(float));
    float*  cnt  = (float*)take((size_t)G * sizeof(float));

    // layer 1: F_IN -> HEADS*HID (concat), ELU
    run_layer(x, F_IN, W1, as1, ad1, b1, HEADS, HID, N, ei, E, Etot,
              xb, Bp, hbuf, als, ald, mbuf, dbuf, ebuf, agg, actA, true, stream);
    // layer 2: HEADS*HID -> HEADS*HID (concat), ELU
    run_layer(actA, HEADS * HID, W2, as2, ad2, b2, HEADS, HID, N, ei, E, Etot,
              xb, Bp, hbuf, als, ald, mbuf, dbuf, ebuf, agg, actA, true, stream);
    // layer 3: HEADS*HID -> OUT, single head (mean over 1 head == identity), no ELU
    run_layer(actA, HEADS * HID, W3, as3, ad3, b3, 1, OUT_C, N, ei, E, Etot,
              xb, Bp, hbuf, als, ald, mbuf, dbuf, ebuf, agg, actA, false, stream);

    // global mean pool
    fill_f32<<<(unsigned)cdivl((long)G * OUT_C, 256), 256, 0, stream>>>(pool, 0.0f, (long)G * OUT_C);
    fill_f32<<<1, 256, 0, stream>>>(cnt, 0.0f, G);
    pool_acc<<<(unsigned)cdivl((long)N * OUT_C, 256), 256, 0, stream>>>(actA, batch, pool, cnt, N, OUT_C);
    pool_div<<<(unsigned)cdivl((long)G * OUT_C, 256), 256, 0, stream>>>(pool, cnt, (float*)d_out, G, OUT_C);
}
